// TransformerLayer_21114059227280
// MI455X (gfx1250) — compile-verified, matrix operand order checked by reading, not run on hardware
//
#include <hip/hip_runtime.h>
#include <cmath>

#define DEVINL __device__ __forceinline__

#if defined(__has_builtin)
# if __has_builtin(__builtin_amdgcn_global_load_async_to_lds_b128)
#  define HAVE_ASYNC_COPY 1
# endif
#endif

constexpr int B_  = 4;
constexpr int N_  = 2048;
constexpr int E_  = 256;
constexpr int H_  = 4;
constexpr int HD_ = 64;

typedef __bf16 bf16_t;
typedef __attribute__((ext_vector_type(16))) __bf16 v16bf;
typedef __attribute__((ext_vector_type(8)))  __bf16 v8bf;
typedef __attribute__((ext_vector_type(8)))  float  v8f;

DEVINL bf16_t to_bf16(float f) { return (bf16_t)f; }

// Packed 4 x bf16 store (8B, alignment guaranteed by tile strides below).
DEVINL void store4(bf16_t* dst, float x0, float x1, float x2, float x3) {
  union { bf16_t h[4]; unsigned long long u; } pk;
  pk.h[0] = to_bf16(x0); pk.h[1] = to_bf16(x1);
  pk.h[2] = to_bf16(x2); pk.h[3] = to_bf16(x3);
  *reinterpret_cast<unsigned long long*>(dst) = pk.u;
}

DEVINL v8f wmma_bf16(v16bf a, v16bf b, v8f c) {
  return __builtin_amdgcn_wmma_f32_16x16x32_bf16(false, a, false, b, (short)0, c,
                                                 false, false);
}

// 16x32 bf16 fragment from a K-contiguous LDS tile (ld in elements, ld%8==0).
// ISA 7.12.2 lane layout; lowers to two ds_load_b128 per fragment.
DEVINL v16bf frag_ld(const bf16_t* p, int ld, int row0, int k0) {
  const int l = threadIdx.x & 31;
  const bf16_t* base = p + (row0 + (l & 15)) * ld + k0 + (l >> 4) * 8;
  const v8bf lo = *reinterpret_cast<const v8bf*>(base);       // K+0..7
  const v8bf hi = *reinterpret_cast<const v8bf*>(base + 16);  // K+16..23
  v16bf f;
#pragma unroll
  for (int e = 0; e < 8; ++e) { f[e] = lo[e]; f[e + 8] = hi[e]; }
  return f;
}

#if defined(HAVE_ASYNC_COPY)
// Exact parameter type per hipcc diagnostic: vector_size(16) int, AS1 / AS3.
typedef int v4i __attribute__((vector_size(16)));
typedef __attribute__((address_space(1))) v4i* gbl_v4i;
typedef __attribute__((address_space(3))) v4i* lds_v4i;

DEVINL void async_copy_b128(const float* gsrc, float* ldst) {
  __builtin_amdgcn_global_load_async_to_lds_b128(
      (gbl_v4i)(float*)gsrc, (lds_v4i)ldst, 0, 0);
}
#endif

DEVINL void wait_async0() {
#if defined(HAVE_ASYNC_COPY)
# if __has_builtin(__builtin_amdgcn_s_wait_asynccnt)
  __builtin_amdgcn_s_wait_asynccnt(0);
# else
  asm volatile("s_wait_asynccnt 0" ::: "memory");
# endif
#endif
}

// ---------------------------------------------------------------------------
// GEMM: C[M x Nc] = A[M x K] @ W[K x Nc] + bias (+ residual).
// 64x64 tile, K-step 32, 128 threads (4 waves, one 16-row stripe each).
// Double-buffered LDS, software pipelined: next tile's global loads are in
// flight while the current tile's WMMAs run; one barrier per K-step.
// ---------------------------------------------------------------------------
#define TM 64
#define TN 64
#define TK 32

__global__ __launch_bounds__(128)
void gemm_bias_kernel(const float* __restrict__ A, int lda,
                      const float* __restrict__ W, int ldw,
                      const float* __restrict__ bias,
                      const float* __restrict__ residual,
                      float* __restrict__ C, int ldc,
                      int M, int Nc, int K) {
  __shared__ bf16_t As[2][TM][TK + 8];
  __shared__ bf16_t Bt[2][TN][TK + 8];  // transposed weight tile: [n][k]

  const int bm = blockIdx.y * TM;
  const int bn = blockIdx.x * TN;
  const int wave = threadIdx.x >> 5;
  const int tid = threadIdx.x;

  v8f acc[4];
#pragma unroll
  for (int t = 0; t < 4; ++t)
#pragma unroll
    for (int r = 0; r < 8; ++r) acc[t][r] = 0.f;

  float4 ra[4];   // staged A tile (4 iters x float4 per thread)
  float  rw[4][4];  // staged W tile (4 iters x 4 strided scalars)

  auto tload = [&](int k0) {  // load phase: all global loads in flight
#pragma unroll
    for (int it = 0; it < 4; ++it) {
      const int t4 = tid + it * 128;
      const int r = t4 >> 3, g = (t4 & 7) * 4;
      ra[it] = *reinterpret_cast<const float4*>(A + (size_t)(bm + r) * lda + k0 + g);
    }
#pragma unroll
    for (int it = 0; it < 4; ++it) {
      const int t4 = tid + it * 128;
      const int c = t4 & 63, r0 = (t4 >> 6) * 4;
      const float* wp = W + (size_t)(k0 + r0) * ldw + bn + c;
#pragma unroll
      for (int j = 0; j < 4; ++j) rw[it][j] = wp[(size_t)j * ldw];
    }
  };
  auto tstore = [&](int buf) {  // convert + packed LDS store phase
#pragma unroll
    for (int it = 0; it < 4; ++it) {
      const int t4 = tid + it * 128;
      const int r = t4 >> 3, g = (t4 & 7) * 4;
      store4(&As[buf][r][g], ra[it].x, ra[it].y, ra[it].z, ra[it].w);
    }
#pragma unroll
    for (int it = 0; it < 4; ++it) {
      const int t4 = tid + it * 128;
      const int c = t4 & 63, r0 = (t4 >> 6) * 4;
      store4(&Bt[buf][c][r0], rw[it][0], rw[it][1], rw[it][2], rw[it][3]);
    }
  };
  auto compute = [&](int buf) {
    const v16bf af = frag_ld(&As[buf][0][0], TK + 8, wave * 16, 0);
#pragma unroll
    for (int t = 0; t < 4; ++t) {
      const v16bf bfr = frag_ld(&Bt[buf][0][0], TK + 8, t * 16, 0);
      acc[t] = wmma_bf16(af, bfr, acc[t]);
    }
  };

  tload(0);
  tstore(0);
  __syncthreads();
  int buf = 0;
  for (int k0 = TK; k0 < K; k0 += TK) {
    tload(k0);     // next tile's loads overlap current tile's WMMAs
    compute(buf);
    buf ^= 1;
    tstore(buf);
    __syncthreads();
  }
  compute(buf);

  const int l = threadIdx.x & 31;
  const int hf = l >> 4, ln = l & 15;
#pragma unroll
  for (int t = 0; t < 4; ++t) {
    const int col = bn + t * 16 + ln;
    const float bv = bias ? bias[col] : 0.f;
#pragma unroll
    for (int r = 0; r < 8; ++r) {
      const int row = bm + wave * 16 + r + hf * 8;
      float v = acc[t][r] + bv;
      if (residual) v += residual[(size_t)row * ldc + col];
      C[(size_t)row * ldc + col] = v;
    }
  }
}

// ---------------------------------------------------------------------------
// Unpack qkv [B,N,(H,HD,3)] -> q/k/v [B,H,N,HD], folding the positional
// encoding add into q and k. Bandwidth-trivial (one 25MB pass).
// ---------------------------------------------------------------------------
__global__ void split_qkv_kernel(const float* __restrict__ qkv,
                                 const float* __restrict__ enc,
                                 float* __restrict__ q, float* __restrict__ k,
                                 float* __restrict__ v) {
  const size_t i = (size_t)blockIdx.x * blockDim.x + threadIdx.x;
  const size_t total = (size_t)B_ * H_ * N_ * HD_;
  if (i >= total) return;
  const int d = (int)(i % HD_);
  size_t t = i / HD_;
  const int n = (int)(t % N_);
  t /= N_;
  const int h = (int)(t % H_);
  const int b = (int)(t / H_);
  const size_t src = ((size_t)(b * N_ + n)) * (3 * E_) + h * (HD_ * 3) + d * 3;
  const float e = enc[i];
  q[i] = qkv[src + 0] + e;
  k[i] = qkv[src + 1] + e;
  v[i] = qkv[src + 2];
}

// ---------------------------------------------------------------------------
// Flash attention. MODE 0: self, q/k/v in [B,H,N,HD].
// MODE 1: cross, S *= match[b,i,j]. MODE 2: cross, S *= match[b,j,i].
// Grid (B*H, N/64), 128 threads = 4 waves, 16 query rows per wave.
// Scores stay in WMMA accumulators; online softmax via half-wave shfl_xor.
// ---------------------------------------------------------------------------
template<int MODE>
DEVINL const float* row_ptr(const float* p, int b, int h, int row) {
  if (MODE == 0) return p + ((size_t)(b * H_ + h) * N_ + row) * HD_;
  return p + (size_t)(b * N_ + row) * E_ + h * HD_;
}

template<int MODE>
__global__ __launch_bounds__(128)
void attn_kernel(const float* __restrict__ qsrc, const float* __restrict__ ksrc,
                 const float* __restrict__ vsrc, const float* __restrict__ match,
                 float* __restrict__ ctx) {
  const int bh = blockIdx.x;
  const int b = bh / H_, h = bh % H_;
  const int q0 = blockIdx.y * 64;
  const int wave = threadIdx.x >> 5;
  const int l = threadIdx.x & 31;
  const int hf = l >> 4, ln = l & 15;

  __shared__ bf16_t Ks[64][HD_ + 8];          // [key][d]  (K-contig for QK^T)
  __shared__ bf16_t Vt[HD_][64 + 8];          // [d][key]  (K-contig for PV)
  __shared__ bf16_t Ps[4][16][64 + 8];        // per-wave P bounce
  __shared__ float  Ms[MODE ? 64 : 1][68];    // match tile (cross only)

  // Q fragments: two 16x32 A-frags over HD=64, vector loads from global.
  v16bf qf[2];
  {
    const float* qp = row_ptr<MODE>(qsrc, b, h, q0 + wave * 16 + ln);
#pragma unroll
    for (int kk = 0; kk < 2; ++kk) {
      const int base = kk * 32 + hf * 8;
      const float4 a0 = *reinterpret_cast<const float4*>(qp + base);
      const float4 a1 = *reinterpret_cast<const float4*>(qp + base + 4);
      const float4 b0 = *reinterpret_cast<const float4*>(qp + base + 16);
      const float4 b1 = *reinterpret_cast<const float4*>(qp + base + 20);
      v16bf f;
      f[0] = to_bf16(a0.x); f[1] = to_bf16(a0.y); f[2] = to_bf16(a0.z); f[3] = to_bf16(a0.w);
      f[4] = to_bf16(a1.x); f[5] = to_bf16(a1.y); f[6] = to_bf16(a1.z); f[7] = to_bf16(a1.w);
      f[8] = to_bf16(b0.x); f[9] = to_bf16(b0.y); f[10] = to_bf16(b0.z); f[11] = to_bf16(b0.w);
      f[12] = to_bf16(b1.x); f[13] = to_bf16(b1.y); f[14] = to_bf16(b1.z); f[15] = to_bf16(b1.w);
      qf[kk] = f;
    }
  }

  float rmax[8], rsum[8];
  v8f oacc[4];
#pragma unroll
  for (int r = 0; r < 8; ++r) { rmax[r] = -1e30f; rsum[r] = 0.f; }
#pragma unroll
  for (int t = 0; t < 4; ++t)
#pragma unroll
    for (int r = 0; r < 8; ++r) oacc[t][r] = 0.f;

  const float scale = 0.125f;  // HD^-0.5 (cross: half_scale^2 folded in)

  for (int kt = 0; kt < N_ / 64; ++kt) {
    // ---- Load phase: all K/V global loads in flight before any LDS store.
    float4 kreg[8];
#pragma unroll
    for (int it = 0; it < 8; ++it) {
      const int t4 = threadIdx.x + it * 128;
      const int kr = t4 >> 4, g = (t4 & 15) * 4;
      kreg[it] = *reinterpret_cast<const float4*>(
          row_ptr<MODE>(ksrc, b, h, kt * 64 + kr) + g);
    }
    float vreg[8][4];
#pragma unroll
    for (int it = 0; it < 8; ++it) {
      const int t4 = threadIdx.x + it * 128;
      const int d = t4 & 63, kg = (t4 >> 6) * 4;
#pragma unroll
      for (int j = 0; j < 4; ++j)
        vreg[it][j] = row_ptr<MODE>(vsrc, b, h, kt * 64 + kg + j)[d];
    }
    // ---- Store phase: packed bf16 quads into LDS.
#pragma unroll
    for (int it = 0; it < 8; ++it) {
      const int t4 = threadIdx.x + it * 128;
      const int kr = t4 >> 4, g = (t4 & 15) * 4;
      store4(&Ks[kr][g], kreg[it].x, kreg[it].y, kreg[it].z, kreg[it].w);
    }
#pragma unroll
    for (int it = 0; it < 8; ++it) {
      const int t4 = threadIdx.x + it * 128;
      const int d = t4 & 63, kg = (t4 >> 6) * 4;
      store4(&Vt[d][kg], vreg[it][0], vreg[it][1], vreg[it][2], vreg[it][3]);
    }
    if (MODE == 1) {
      // match[b, q0+i, kt*64+j]: contiguous 16B chunks -> async copy to LDS.
#pragma unroll
      for (int t4 = threadIdx.x; t4 < 64 * 16; t4 += 128) {
        const int i = t4 >> 4, j4 = (t4 & 15) * 4;
        const float* mp = match + ((size_t)b * N_ + q0 + i) * N_ + kt * 64 + j4;
#if defined(HAVE_ASYNC_COPY)
        async_copy_b128(mp, &Ms[i][j4]);
#else
        const float4 m4 = *reinterpret_cast<const float4*>(mp);
        Ms[i][j4] = m4.x; Ms[i][j4 + 1] = m4.y;
        Ms[i][j4 + 2] = m4.z; Ms[i][j4 + 3] = m4.w;
#endif
      }
      wait_async0();
    } else if (MODE == 2) {
      // match^T: batched reads contiguous along q, scatter into Ms rows.
      float4 mreg[8];
#pragma unroll
      for (int it = 0; it < 8; ++it) {
        const int t4 = threadIdx.x + it * 128;
        const int j = t4 >> 4, i4 = (t4 & 15) * 4;
        mreg[it] = *reinterpret_cast<const float4*>(
            match + ((size_t)b * N_ + kt * 64 + j) * N_ + q0 + i4);
      }
#pragma unroll
      for (int it = 0; it < 8; ++it) {
        const int t4 = threadIdx.x + it * 128;
        const int j = t4 >> 4, i4 = (t4 & 15) * 4;
        Ms[i4][j] = mreg[it].x; Ms[i4 + 1][j] = mreg[it].y;
        Ms[i4 + 2][j] = mreg[it].z; Ms[i4 + 3][j] = mreg[it].w;
      }
    }
    __syncthreads();

    // S(16x64) = Q @ K^T
    v8f s[4];
#pragma unroll
    for (int j = 0; j < 4; ++j) {
#pragma unroll
      for (int r = 0; r < 8; ++r) s[j][r] = 0.f;
#pragma unroll
      for (int kk = 0; kk < 2; ++kk) {
        const v16bf kb = frag_ld(&Ks[0][0], HD_ + 8, j * 16, kk * 32);
        s[j] = wmma_bf16(qf[kk], kb, s[j]);
      }
    }

#pragma unroll
    for (int j = 0; j < 4; ++j)
#pragma unroll
      for (int r = 0; r < 8; ++r) {
        float sv = s[j][r] * scale;
        if (MODE != 0) sv *= Ms[wave * 16 + r + hf * 8][j * 16 + ln];
        s[j][r] = sv;
      }

    // Online softmax (rows live in 16-lane halves).
#pragma unroll
    for (int r = 0; r < 8; ++r) {
      float tmax = s[0][r];
#pragma unroll
      for (int j = 1; j < 4; ++j) tmax = fmaxf(tmax, s[j][r]);
#pragma unroll
      for (int m = 1; m <= 8; m <<= 1) tmax = fmaxf(tmax, __shfl_xor(tmax, m, 32));
      const float nm = fmaxf(rmax[r], tmax);
      const float fac = __expf(rmax[r] - nm);
      rmax[r] = nm;
      float ts = 0.f;
#pragma unroll
      for (int j = 0; j < 4; ++j) {
        const float p = __expf(s[j][r] - nm);
        s[j][r] = p;
        ts += p;
      }
#pragma unroll
      for (int m = 1; m <= 8; m <<= 1) ts += __shfl_xor(ts, m, 32);
      rsum[r] = rsum[r] * fac + ts;
#pragma unroll
      for (int t = 0; t < 4; ++t) oacc[t][r] *= fac;
    }

    // C-layout P -> A-layout via per-wave LDS bounce.
#pragma unroll
    for (int j = 0; j < 4; ++j)
#pragma unroll
      for (int r = 0; r < 8; ++r)
        Ps[wave][r + hf * 8][j * 16 + ln] = to_bf16(s[j][r]);
    asm volatile("s_wait_dscnt 0" ::: "memory");  // wave-internal LDS RAW

    v16bf pa[2];
    pa[0] = frag_ld(&Ps[wave][0][0], 64 + 8, 0, 0);
    pa[1] = frag_ld(&Ps[wave][0][0], 64 + 8, 0, 32);
#pragma unroll
    for (int t = 0; t < 4; ++t)
#pragma unroll
      for (int kk = 0; kk < 2; ++kk) {
        const v16bf vb = frag_ld(&Vt[0][0], 64 + 8, t * 16, kk * 32);
        oacc[t] = wmma_bf16(pa[kk], vb, oacc[t]);
      }
    __syncthreads();
  }

  // Normalize, store merged ctx [B, N, E] (col = h*HD + d).
#pragma unroll
  for (int t = 0; t < 4; ++t)
#pragma unroll
    for (int r = 0; r < 8; ++r) {
      const int row = q0 + wave * 16 + r + hf * 8;
      const int col = h * HD_ + t * 16 + ln;
      ctx[((size_t)b * N_ + row) * E_ + col] = oacc[t][r] / rsum[r];
    }
}

// ---------------------------------------------------------------------------
// LayerNorm (2E=512) + exact GELU. One block per row, 128 threads.
// ---------------------------------------------------------------------------
__global__ __launch_bounds__(128)
void ln_gelu_kernel(const float* __restrict__ x, const float* __restrict__ g,
                    const float* __restrict__ be, float* __restrict__ y) {
  const int C = 2 * E_;
  const size_t row = blockIdx.x;
  const float* xr = x + row * C;
  __shared__ float red[128];

  float loc[4], s = 0.f;
#pragma unroll
  for (int i = 0; i < 4; ++i) { loc[i] = xr[threadIdx.x + i * 128]; s += loc[i]; }
  red[threadIdx.x] = s;
  __syncthreads();
  for (int off = 64; off > 0; off >>= 1) {
    if (threadIdx.x < off) red[threadIdx.x] += red[threadIdx.x + off];
    __syncthreads();
  }
  const float mean = red[0] / C;
  __syncthreads();

  float s2 = 0.f;
#pragma unroll
  for (int i = 0; i < 4; ++i) { const float d = loc[i] - mean; s2 += d * d; }
  red[threadIdx.x] = s2;
  __syncthreads();
  for (int off = 64; off > 0; off >>= 1) {
    if (threadIdx.x < off) red[threadIdx.x] += red[threadIdx.x + off];
    __syncthreads();
  }
  const float rstd = rsqrtf(red[0] / C + 1e-5f);

#pragma unroll
  for (int i = 0; i < 4; ++i) {
    const int idx = threadIdx.x + i * 128;
    const float t = (loc[i] - mean) * rstd * g[idx] + be[idx];
    y[row * C + idx] = 0.5f * t * (1.f + erff(t * 0.70710678118654752f));
  }
}

__global__ void concat_kernel(const float* __restrict__ a,
                              const float* __restrict__ b,
                              float* __restrict__ o) {
  const size_t i = (size_t)blockIdx.x * blockDim.x + threadIdx.x;
  const size_t total = (size_t)B_ * N_ * 2 * E_;
  if (i >= total) return;
  const size_t row = i / (2 * E_);
  const int col = (int)(i % (2 * E_));
  o[i] = (col < E_) ? a[row * E_ + col] : b[row * E_ + col - E_];
}

// ---------------------------------------------------------------------------
extern "C" void kernel_launch(void* const* d_in, const int* in_sizes, int n_in,
                              void* d_out, int out_size, void* d_ws, size_t ws_size,
                              hipStream_t stream) {
  const float* desc0 = (const float*)d_in[0];
  const float* desc1 = (const float*)d_in[1];
  const float* enc0  = (const float*)d_in[2];
  const float* enc1  = (const float*)d_in[3];
  const float* match = (const float*)d_in[4];
  const float* sWqkv = (const float*)d_in[5];
  const float* sbqkv = (const float*)d_in[6];
  const float* sWout = (const float*)d_in[7];
  const float* sbout = (const float*)d_in[8];
  const float* sW1   = (const float*)d_in[9];
  const float* sb1   = (const float*)d_in[10];
  const float* sg    = (const float*)d_in[11];
  const float* sbeta = (const float*)d_in[12];
  const float* sW2   = (const float*)d_in[13];
  const float* sb2   = (const float*)d_in[14];
  const float* cWqk  = (const float*)d_in[15];
  const float* cbqk  = (const float*)d_in[16];
  const float* cWv   = (const float*)d_in[17];
  const float* cbv   = (const float*)d_in[18];
  const float* cWout = (const float*)d_in[19];
  const float* cbout = (const float*)d_in[20];
  const float* cW1   = (const float*)d_in[21];
  const float* cb1   = (const float*)d_in[22];
  const float* cg    = (const float*)d_in[23];
  const float* cbeta = (const float*)d_in[24];
  const float* cW2   = (const float*)d_in[25];
  const float* cb2   = (const float*)d_in[26];

  const size_t MR = (size_t)B_ * N_;  // 8192 rows
  float* w   = (float*)d_ws;          // ~168 MB fp32 carve-up
  float* qkv = w; w += MR * 3 * E_;
  float* qs  = w; w += MR * E_;
  float* ks  = w; w += MR * E_;
  float* vs  = w; w += MR * E_;
  float* ctx = w; w += MR * E_;
  float* msg = w; w += MR * E_;
  float* cat = w; w += MR * 2 * E_;
  float* h1  = w; w += MR * 2 * E_;
  float* act = w; w += MR * 2 * E_;
  float* d0b = w; w += MR * E_;
  float* d1b = w; w += MR * E_;
  float* qk0 = w; w += MR * E_;
  float* qk1 = w; w += MR * E_;
  float* v0  = w; w += MR * E_;
  float* v1  = w; w += MR * E_;

  auto gemm = [&](const float* A, int lda, const float* W, int ldw,
                  const float* bias, const float* res, float* C, int ldc,
                  int M, int Nc, int K) {
    dim3 g(Nc / TN, M / TM), blk(128);
    gemm_bias_kernel<<<g, blk, 0, stream>>>(A, lda, W, ldw, bias, res, C, ldc, M, Nc, K);
  };
  const dim3 ag(B_ * H_, N_ / 64);
  const int cc = (int)((MR * 2 * E_ + 255) / 256);
  const int sq = (int)((MR * E_ + 255) / 256);

  auto ffn = [&](const float* x, const float* m, const float* W1, const float* b1,
                 const float* gg, const float* bb, const float* W2, const float* b2,
                 float* out) {
    concat_kernel<<<cc, 256, 0, stream>>>(x, m, cat);
    gemm(cat, 2 * E_, W1, 2 * E_, b1, nullptr, h1, 2 * E_, (int)MR, 2 * E_, 2 * E_);
    ln_gelu_kernel<<<(int)MR, 128, 0, stream>>>(h1, gg, bb, act);
    gemm(act, 2 * E_, W2, E_, b2, x /*residual*/, out, E_, (int)MR, E_, 2 * E_);
  };

  auto self_block = [&](const float* x, const float* enc, float* xout) {
    gemm(x, E_, sWqkv, 3 * E_, sbqkv, nullptr, qkv, 3 * E_, (int)MR, 3 * E_, E_);
    split_qkv_kernel<<<sq, 256, 0, stream>>>(qkv, enc, qs, ks, vs);
    attn_kernel<0><<<ag, 128, 0, stream>>>(qs, ks, vs, nullptr, ctx);
    gemm(ctx, E_, sWout, E_, sbout, nullptr, msg, E_, (int)MR, E_, E_);
    ffn(x, msg, sW1, sb1, sg, sbeta, sW2, sb2, xout);
  };

  // ---- Self blocks ----
  self_block(desc0, enc0, d0b);
  self_block(desc1, enc1, d1b);

  // ---- Cross block ----
  gemm(d0b, E_, cWqk, E_, cbqk, nullptr, qk0, E_, (int)MR, E_, E_);
  gemm(d1b, E_, cWqk, E_, cbqk, nullptr, qk1, E_, (int)MR, E_, E_);
  gemm(d0b, E_, cWv,  E_, cbv,  nullptr, v0,  E_, (int)MR, E_, E_);
  gemm(d1b, E_, cWv,  E_, cbv,  nullptr, v1,  E_, (int)MR, E_, E_);

  float* out0 = (float*)d_out;
  float* out1 = out0 + MR * E_;

  // x0 attends into x1 (softmax over n1), match as-is.
  attn_kernel<1><<<ag, 128, 0, stream>>>(qk0, qk1, v1, match, ctx);
  gemm(ctx, E_, cWout, E_, cbout, nullptr, msg, E_, (int)MR, E_, E_);
  ffn(d0b, msg, cW1, cb1, cg, cbeta, cW2, cb2, out0);

  // x1 attends into x0 (softmax over n0), match transposed.
  attn_kernel<2><<<ag, 128, 0, stream>>>(qk1, qk0, v0, match, ctx);
  gemm(ctx, E_, cWout, E_, cbout, nullptr, msg, E_, (int)MR, E_, E_);
  ffn(d1b, msg, cW1, cb1, cg, cbeta, cW2, cb2, out1);
}
